// RegionProposalNetwork_7808250544639
// MI455X (gfx1250) — matrix operand order repaired and checked by reading, not compiled, not run on hardware
//
#include <hip/hip_runtime.h>
#include <hip/hip_bf16.h>

// ---------------------------------------------------------------------------
// RPN forward for MI455X (gfx1250, wave32, WMMA).
// conv3x3 512->512 as implicit GEMM (M=512,N=1900,K=4608) with
// v_wmma_f32_16x16x32_f16, 16x64 tile per wave (4 accumulators, A reuse x4),
// branch-free B loads via a zero page; 1x1 convs as small WMMA GEMMs; then
// decode, exact O(N^2) stable-rank top-6000, bitmask NMS, compaction.
// ---------------------------------------------------------------------------

typedef __attribute__((ext_vector_type(16))) _Float16 v16h;
typedef __attribute__((ext_vector_type(8)))  _Float16 v8h;
typedef __attribute__((ext_vector_type(8)))  float    v8f;

#define NEG_INF (-__builtin_inff())

// Problem constants (fixed by the reference)
#define NB   2
#define HH   38
#define WW   50
#define PP   1900          // HH*WW
#define CC   512
#define KTOT 4608          // CC*9
#define AA   17100         // PP*9 anchors per batch
#define NPRE 6000
#define NPOST 300
#define NW   188           // ceil(6000/32) suppression words per row

// d_out layout (floats)
#define OFF_LOCS   0
#define OFF_SCORES 136800   // 2*17100*4
#define OFF_ROIS   205200   // +2*17100*2
#define OFF_RIDX   207600   // +600*4
#define OFF_ANCH   208200   // +600
#define OUT_TOTAL  276600

// ---------------------------------------------------------------------------
// Packing kernels
// ---------------------------------------------------------------------------
__global__ void pack_weights_k(const float* __restrict__ cw,
                               const float* __restrict__ lw,
                               const float* __restrict__ sw,
                               _Float16* __restrict__ wt,
                               _Float16* __restrict__ wl,
                               _Float16* __restrict__ wsc,
                               _Float16* __restrict__ zbuf) {
  long i = (long)blockIdx.x * blockDim.x + threadIdx.x;
  if (i < (long)CC * KTOT) {
    int oc  = (int)(i / KTOT);
    int k   = (int)(i % KTOT);
    int tap = k / CC;
    int ic  = k % CC;
    wt[i] = (_Float16)cw[((size_t)oc * CC + ic) * 9 + tap];  // OIHW, tap = ky*3+kx
  }
  if (i < 48 * 512) {
    int oc = (int)(i / 512), ic = (int)(i % 512);
    wl[i] = (oc < 36) ? (_Float16)lw[oc * 512 + ic] : (_Float16)0.0f;
  }
  if (i < 32 * 512) {
    int oc = (int)(i / 512), ic = (int)(i % 512);
    wsc[i] = (oc < 18) ? (_Float16)sw[oc * 512 + ic] : (_Float16)0.0f;
  }
  if (i < 512) zbuf[i] = (_Float16)0.0f;   // zero page for OOB B fragments
}

__global__ void pack_input_k(const float* __restrict__ x, _Float16* __restrict__ xt) {
  long i = (long)blockIdx.x * blockDim.x + threadIdx.x;
  if (i >= (long)NB * PP * CC) return;
  int b   = (int)(i / ((long)PP * CC));
  int rem = (int)(i % ((long)PP * CC));
  int p   = rem / CC;
  int ic  = rem % CC;
  // NCHW -> [b][p][ic]
  xt[i] = (_Float16)x[((size_t)b * CC + ic) * PP + p];
}

// ---------------------------------------------------------------------------
// Anchors + output init (rois zero-pad, roi_indices)
// ---------------------------------------------------------------------------
__global__ void init_anchors_k(float* __restrict__ out) {
  int i = blockIdx.x * blockDim.x + threadIdx.x;
  if (i < AA) {
    int p = i / 9, a = i % 9;
    int y = p / WW, x = p % WW;
    const float ratios[3] = {0.5f, 1.0f, 2.0f};
    const float scales[3] = {8.0f, 16.0f, 32.0f};
    float r = ratios[a / 3], s = scales[a % 3];
    float h = 16.0f * s * sqrtf(r);
    float w = 16.0f * s * sqrtf(1.0f / r);
    float sy = y * 16.0f, sx = x * 16.0f;
    float* an = out + OFF_ANCH + (size_t)i * 4;
    an[0] = sy + 8.0f - 0.5f * h;
    an[1] = sx + 8.0f - 0.5f * w;
    an[2] = sy + 8.0f + 0.5f * h;
    an[3] = sx + 8.0f + 0.5f * w;
  }
  if (i < NB * NPOST * 4) out[OFF_ROIS + i] = 0.0f;
  if (i < NB * NPOST)     out[OFF_RIDX + i] = (float)(i / NPOST);
}

// ---------------------------------------------------------------------------
// 3x3 conv as implicit GEMM, one wave per 16x64 output tile (4 N-columns,
// A fragment reused 4x). OOB B columns read a zero page -> no exec masking
// in the hot loop. D -> +bias -> relu -> f16 ht[b][p][oc] channel-contiguous.
// ---------------------------------------------------------------------------
__global__ __launch_bounds__(32) void conv3x3_wmma_k(const _Float16* __restrict__ wt,
                                                     const _Float16* __restrict__ xt,
                                                     const float* __restrict__ bias,
                                                     const _Float16* __restrict__ zbuf,
                                                     _Float16* __restrict__ ht) {
  const int mt = blockIdx.x;          // 0..31 output-channel tile
  const int nt = blockIdx.y;          // 0..29 position macro-tile (64 wide)
  const int b  = blockIdx.z;
  const int lane = threadIdx.x;
  const int hx = lane >> 4;           // 0/1 half-wave
  const int lm = lane & 15;
  const int oc = mt * 16 + lm;        // A row (M) for this lane
  const _Float16* wrow = wt + (size_t)oc * KTOT;
  const _Float16* xb   = xt + (size_t)b * PP * CC;

  int  p[4], yy[4], xv[4];
  bool pv[4];
#pragma unroll
  for (int c = 0; c < 4; ++c) {
    p[c]  = nt * 64 + c * 16 + lm;    // B/D column (N) for this lane
    pv[c] = p[c] < PP;
    yy[c] = p[c] / WW;
    xv[c] = p[c] - yy[c] * WW;
  }

  v8f acc0 = {}, acc1 = {}, acc2 = {}, acc3 = {};
  for (int tap = 0; tap < 9; ++tap) {
    const int dy = tap / 3 - 1, dx = tap % 3 - 1;
    const _Float16* brow[4];
#pragma unroll
    for (int c = 0; c < 4; ++c) {
      const int iy = yy[c] + dy, ix = xv[c] + dx;
      const bool bv = pv[c] && (unsigned)iy < (unsigned)HH && (unsigned)ix < (unsigned)WW;
      brow[c] = bv ? (xb + (size_t)(iy * WW + ix) * CC) : zbuf;
      __builtin_prefetch(brow[c], 0, 1);             // global_prefetch_b8
    }
    for (int kc = 0; kc < CC; kc += 32) {
      const int k0 = tap * CC + kc;
      v16h av;
      // A 16x32, contiguous k order chosen by this code (not the ISA table): lanes<16 K=0..15, lanes>=16 K=16..31
      av = *(const v16h*)(wrow + k0 + hx * 16);
      // B 32x16, same contiguous k order as A: lanes<16 K=0..15, lanes>=16 K=16..31
      v16h b0 = *(const v16h*)(brow[0] + kc + hx * 16);
      v16h b1 = *(const v16h*)(brow[1] + kc + hx * 16);
      v16h b2 = *(const v16h*)(brow[2] + kc + hx * 16);
      v16h b3 = *(const v16h*)(brow[3] + kc + hx * 16);
      acc0 = __builtin_amdgcn_wmma_f32_16x16x32_f16(false, av, false, b0, (short)0, acc0, false, false);
      acc1 = __builtin_amdgcn_wmma_f32_16x16x32_f16(false, av, false, b1, (short)0, acc1, false, false);
      acc2 = __builtin_amdgcn_wmma_f32_16x16x32_f16(false, av, false, b2, (short)0, acc2, false, false);
      acc3 = __builtin_amdgcn_wmma_f32_16x16x32_f16(false, av, false, b3, (short)0, acc3, false, false);
    }
  }
  // D: VGPR v -> M = v + 8*hx, N = lm -> contiguous 8 channels per lane
  const int ocb = mt * 16 + hx * 8;
  float bl[8];
#pragma unroll
  for (int v = 0; v < 8; ++v) bl[v] = bias[ocb + v];
  v8f accs[4] = {acc0, acc1, acc2, acc3};
#pragma unroll
  for (int c = 0; c < 4; ++c) {
    if (pv[c]) {
      _Float16* orow = ht + ((size_t)b * PP + p[c]) * CC + ocb;
      v8h o;
#pragma unroll
      for (int v = 0; v < 8; ++v)
        o[v] = (_Float16)fmaxf(accs[c][v] + bl[v], 0.0f);
      *(v8h*)orow = o;
    }
  }
}

// ---------------------------------------------------------------------------
// 1x1 conv GEMM: K=512, padded-M packed weights, fp32 output [b][p][Mreal].
// ---------------------------------------------------------------------------
__global__ __launch_bounds__(32) void conv1x1_wmma_k(const _Float16* __restrict__ wpk,
                                                     const int Mreal,
                                                     const _Float16* __restrict__ ht,
                                                     const _Float16* __restrict__ zbuf,
                                                     const float* __restrict__ bias,
                                                     float* __restrict__ outr) {
  const int mt = blockIdx.x, nt = blockIdx.y, b = blockIdx.z;
  const int lane = threadIdx.x;
  const int hx = lane >> 4, lm = lane & 15;
  const int oc = mt * 16 + lm;
  const int p  = nt * 16 + lm;
  const bool pv = p < PP;
  const _Float16* wrow = wpk + (size_t)oc * CC;
  const _Float16* brow = pv ? (ht + ((size_t)b * PP + p) * CC) : zbuf;
  v8f acc = {};
  for (int kc = 0; kc < CC; kc += 32) {
    v16h av;
    av = *(const v16h*)(wrow + kc + hx * 16);
    v16h bv16 = *(const v16h*)(brow + kc + hx * 16);
    acc = __builtin_amdgcn_wmma_f32_16x16x32_f16(false, av, false, bv16,
                                                 (short)0, acc, false, false);
  }
  if (pv) {
#pragma unroll
    for (int v = 0; v < 8; ++v) {
      const int ocw = mt * 16 + hx * 8 + v;
      if (ocw < Mreal)
        outr[((size_t)b * PP + p) * Mreal + ocw] = acc[v] + bias[ocw];
    }
  }
}

// ---------------------------------------------------------------------------
// Decode: write rpn_locs / rpn_scores outputs, loc2bbox + clip + min-size,
// fg score via softmax-over-2 (= sigmoid(s1-s0)), -inf mask for invalid.
// ---------------------------------------------------------------------------
__global__ void decode_k(const float* __restrict__ locraw,
                         const float* __restrict__ scoreraw,
                         const int* __restrict__ ihp, const int* __restrict__ iwp,
                         float* __restrict__ roibuf, float* __restrict__ scorebuf,
                         float* __restrict__ out) {
  int idx = blockIdx.x * blockDim.x + threadIdx.x;
  if (idx >= NB * AA) return;
  int b = idx / AA, t = idx % AA;
  int p = t / 9, a = t % 9;
  const float* lr = locraw + ((size_t)b * PP + p) * 36 + a * 4;
  float l0 = lr[0], l1 = lr[1], l2 = lr[2], l3 = lr[3];
  const float* sr = scoreraw + ((size_t)b * PP + p) * 18 + a * 2;
  float s0 = sr[0], s1 = sr[1];
  // raw outputs (NHWC flattening matches t = p*9 + a)
  out[OFF_LOCS   + (size_t)idx * 4 + 0] = l0;
  out[OFF_LOCS   + (size_t)idx * 4 + 1] = l1;
  out[OFF_LOCS   + (size_t)idx * 4 + 2] = l2;
  out[OFF_LOCS   + (size_t)idx * 4 + 3] = l3;
  out[OFF_SCORES + (size_t)idx * 2 + 0] = s0;
  out[OFF_SCORES + (size_t)idx * 2 + 1] = s1;
  const float* an = out + OFF_ANCH + (size_t)t * 4;
  float ah = an[2] - an[0], aw = an[3] - an[1];
  float cy = an[0] + 0.5f * ah, cx = an[1] + 0.5f * aw;
  float cy2 = l0 * ah + cy, cx2 = l1 * aw + cx;
  float h2 = expf(l2) * ah, w2 = expf(l3) * aw;
  float fh = (float)ihp[0], fw = (float)iwp[0];
  float r0 = fminf(fmaxf(cy2 - 0.5f * h2, 0.0f), fh);
  float r1 = fminf(fmaxf(cx2 - 0.5f * w2, 0.0f), fw);
  float r2 = fminf(fmaxf(cy2 + 0.5f * h2, 0.0f), fh);
  float r3 = fminf(fmaxf(cx2 + 0.5f * w2, 0.0f), fw);
  bool valid = (r2 - r0 >= 16.0f) && (r3 - r1 >= 16.0f);
  float fg = 1.0f / (1.0f + expf(s0 - s1));       // softmax component 1
  float* rb = roibuf + (size_t)idx * 4;
  rb[0] = r0; rb[1] = r1; rb[2] = r2; rb[3] = r3;
  scorebuf[idx] = valid ? fg : NEG_INF;
}

// ---------------------------------------------------------------------------
// Exact stable rank top-6000 (reproduces jax.lax.top_k ordering/ties).
// ---------------------------------------------------------------------------
__global__ void rank_topk_k(const float* __restrict__ scorebuf,
                            const float* __restrict__ roibuf,
                            float* __restrict__ sscore, float* __restrict__ sbox) {
  const int b = blockIdx.y;
  const int i = blockIdx.x * blockDim.x + threadIdx.x;
  const float* s = scorebuf + (size_t)b * AA;
  const float si = (i < AA) ? s[i] : 0.0f;
  int rank = 0;
  __shared__ float cache[256];
  for (int jt = 0; jt < AA; jt += 256) {
    int j = jt + threadIdx.x;
    cache[threadIdx.x] = (j < AA) ? s[j] : 0.0f;
    __syncthreads();
    int lim = AA - jt; if (lim > 256) lim = 256;
    for (int k = 0; k < lim; ++k) {
      float sj = cache[k];
      int j2 = jt + k;
      if (sj > si || (sj == si && j2 < i)) rank++;
    }
    __syncthreads();
  }
  if (i < AA && rank < NPRE) {
    sscore[(size_t)b * NPRE + rank] = si;
    const float* rb = roibuf + ((size_t)b * AA + i) * 4;
    float* db = sbox + ((size_t)b * NPRE + rank) * 4;
    db[0] = rb[0]; db[1] = rb[1]; db[2] = rb[2]; db[3] = rb[3];
  }
}

// ---------------------------------------------------------------------------
// Suppression bit matrix: bit set iff j>i && IoU(i,j) > 0.7.
// ---------------------------------------------------------------------------
__global__ void iou_mask_k(const float* __restrict__ sbox, unsigned* __restrict__ sup) {
  long idx = (long)blockIdx.x * blockDim.x + threadIdx.x;
  if (idx >= (long)NB * NPRE * NW) return;
  int w = (int)(idx % NW);
  int i = (int)((idx / NW) % NPRE);
  int b = (int)(idx / ((long)NW * NPRE));
  const float* bi = sbox + ((size_t)b * NPRE + i) * 4;
  float y0 = bi[0], x0 = bi[1], y1 = bi[2], x1 = bi[3];
  float ai = (y1 - y0) * (x1 - x0);
  unsigned bits = 0u;
  int j0 = w * 32;
  for (int jj = 0; jj < 32; ++jj) {
    int j = j0 + jj;
    if (j <= i || j >= NPRE) continue;
    const float* bj = sbox + ((size_t)b * NPRE + j) * 4;
    float ty = fmaxf(y0, bj[0]), tx = fmaxf(x0, bj[1]);
    float by = fminf(y1, bj[2]), bx = fminf(x1, bj[3]);
    float inter = fmaxf(by - ty, 0.0f) * fmaxf(bx - tx, 0.0f);
    float aj = (bj[2] - bj[0]) * (bj[3] - bj[1]);
    float iou = inter / (ai + aj - inter);
    if (iou > 0.7f) bits |= (1u << jj);
  }
  sup[idx] = bits;
}

// ---------------------------------------------------------------------------
// Sequential greedy scan (one block per batch) + first-300 compaction.
// ---------------------------------------------------------------------------
__global__ void nms_scan_k(const float* __restrict__ sbox,
                           const float* __restrict__ sscore,
                           const unsigned* __restrict__ sup,
                           float* __restrict__ out) {
  const int b = blockIdx.x;
  const int tid = threadIdx.x;
  __shared__ unsigned remv[NW];
  __shared__ int flag;
  __shared__ int outc;
  if (tid < NW) remv[tid] = 0u;
  if (tid == 0) outc = 0;
  __syncthreads();
  for (int i = 0; i < NPRE; ++i) {
    if (tid == 0) {
      bool alive = (((remv[i >> 5] >> (i & 31)) & 1u) == 0u) &&
                   (sscore[(size_t)b * NPRE + i] > NEG_INF);
      flag = alive ? 1 : 0;
      if (alive && outc < NPOST) {
        const float* bx = sbox + ((size_t)b * NPRE + i) * 4;
        float* dst = out + OFF_ROIS + ((size_t)b * NPOST + outc) * 4;
        dst[0] = bx[0]; dst[1] = bx[1]; dst[2] = bx[2]; dst[3] = bx[3];
        outc++;
      }
    }
    __syncthreads();
    if (flag && tid < NW)
      remv[tid] |= sup[((size_t)b * NPRE + i) * NW + tid];
    __syncthreads();
  }
}

// ---------------------------------------------------------------------------
extern "C" void kernel_launch(void* const* d_in, const int* in_sizes, int n_in,
                              void* d_out, int out_size, void* d_ws, size_t ws_size,
                              hipStream_t stream) {
  const float* x       = (const float*)d_in[0];
  const float* conv1_w = (const float*)d_in[1];
  const float* conv1_b = (const float*)d_in[2];
  const float* score_w = (const float*)d_in[3];
  const float* score_b = (const float*)d_in[4];
  const float* loc_w   = (const float*)d_in[5];
  const float* loc_b   = (const float*)d_in[6];
  const int*   img_h   = (const int*)d_in[7];
  const int*   img_w   = (const int*)d_in[8];
  float* out = (float*)d_out;

  char* wsp = (char*)d_ws;
  auto alloc = [&](size_t bytes) -> char* {
    char* r = wsp;
    wsp += (bytes + 255) & ~(size_t)255;
    return r;
  };
  _Float16* wt       = (_Float16*)alloc((size_t)CC * KTOT * 2);     // 4.72 MB
  _Float16* wl       = (_Float16*)alloc((size_t)48 * 512 * 2);
  _Float16* wsc      = (_Float16*)alloc((size_t)32 * 512 * 2);
  _Float16* zbuf     = (_Float16*)alloc((size_t)512 * 2);           // zero page
  _Float16* xt       = (_Float16*)alloc((size_t)NB * PP * CC * 2);  // 3.89 MB
  _Float16* ht       = (_Float16*)alloc((size_t)NB * PP * CC * 2);  // 3.89 MB
  float*    locraw   = (float*)alloc((size_t)NB * PP * 36 * 4);
  float*    scoreraw = (float*)alloc((size_t)NB * PP * 18 * 4);
  float*    roibuf   = (float*)alloc((size_t)NB * AA * 4 * 4);
  float*    scorebuf = (float*)alloc((size_t)NB * AA * 4);
  float*    sbox     = (float*)alloc((size_t)NB * NPRE * 4 * 4);
  float*    sscore   = (float*)alloc((size_t)NB * NPRE * 4);
  unsigned* sup      = (unsigned*)alloc((size_t)NB * NPRE * NW * 4); // 9.0 MB

  // 1) pack weights + input to f16 WMMA-friendly layouts (+ zero page)
  {
    long n = (long)CC * KTOT;
    pack_weights_k<<<(unsigned)((n + 255) / 256), 256, 0, stream>>>(
        conv1_w, loc_w, score_w, wt, wl, wsc, zbuf);
    long m = (long)NB * PP * CC;
    pack_input_k<<<(unsigned)((m + 255) / 256), 256, 0, stream>>>(x, xt);
  }
  // 2) anchors + rois zero-pad + roi_indices
  init_anchors_k<<<(AA + 255) / 256, 256, 0, stream>>>(out);
  // 3) conv3x3 + relu (WMMA implicit GEMM, 16x64 per wave)
  conv3x3_wmma_k<<<dim3(CC / 16, (PP + 63) / 64, NB), 32, 0, stream>>>(
      wt, xt, conv1_b, zbuf, ht);
  // 4) 1x1 convs (WMMA GEMM): loc (36 ch, 3 M-tiles), score (18 ch, 2 M-tiles)
  conv1x1_wmma_k<<<dim3(3, (PP + 15) / 16, NB), 32, 0, stream>>>(
      wl, 36, ht, zbuf, loc_b, locraw);
  conv1x1_wmma_k<<<dim3(2, (PP + 15) / 16, NB), 32, 0, stream>>>(
      wsc, 18, ht, zbuf, score_b, scoreraw);
  // 5) decode + raw outputs
  decode_k<<<(NB * AA + 255) / 256, 256, 0, stream>>>(
      locraw, scoreraw, img_h, img_w, roibuf, scorebuf, out);
  // 6) exact stable top-6000 per batch
  rank_topk_k<<<dim3((AA + 255) / 256, NB), 256, 0, stream>>>(
      scorebuf, roibuf, sscore, sbox);
  // 7) suppression bit matrix
  {
    long n = (long)NB * NPRE * NW;
    iou_mask_k<<<(unsigned)((n + 255) / 256), 256, 0, stream>>>(sbox, sup);
  }
  // 8) greedy NMS scan + compaction into rois
  nms_scan_k<<<NB, 224, 0, stream>>>(sbox, sscore, sup, out);
}
